// HeteGAT_50757923504417
// MI455X (gfx1250) — matrix operand order repaired and checked by reading, not compiled, not run on hardware
//
#include <hip/hip_runtime.h>
#include <hip/hip_bf16.h>
#include <math.h>

// ---------------------------------------------------------------------------
// CDNA5 (gfx1250) WMMA / vector types
// ---------------------------------------------------------------------------
typedef _Float16 v16h_t __attribute__((ext_vector_type(16)));
typedef _Float16 v8h_t  __attribute__((ext_vector_type(8)));
typedef float    v8f_t  __attribute__((ext_vector_type(8)));
typedef float    v4f_t  __attribute__((ext_vector_type(4)));

__device__ __forceinline__ float gelu_exact(float v) {
    return 0.5f * v * (1.0f + erff(v * 0.70710678118654752440f));
}

// ---------------------------------------------------------------------------
// Transpose + f32->f16 convert + zero-pad:  Bt[n, kp] = B[k, n]  (kp = K padded to 32)
// ---------------------------------------------------------------------------
__global__ void convert_bt_kernel(const float* __restrict__ B,
                                  _Float16* __restrict__ Bt,
                                  int K, int N, int Kp)
{
    const size_t i = (size_t)blockIdx.x * blockDim.x + threadIdx.x;
    if (i >= (size_t)N * Kp) return;
    const int n = (int)(i / Kp);
    const int k = (int)(i % Kp);
    Bt[i] = (k < K) ? (_Float16)B[(size_t)k * N + n] : (_Float16)0.f;
}

// ---------------------------------------------------------------------------
// WMMA GEMM: C[M,N] = act(A[M,K] @ B[K,N] + bias[N])
//   B is pre-transposed half:  Bt[N, Kp]  (zero padded K -> Kp, Kp % 32 == 0)
//   act: 0 = none, 1 = leaky_relu(0.01), 2 = exact GELU
// One wave32 computes a 16x64 C tile (4 WMMA accumulators, A fragment reused 4x).
// N must be a multiple of 64 (all call sites: 128 / 512).
// ---------------------------------------------------------------------------
__global__ void wmma_gemm_kernel(const float* __restrict__ A,
                                 const _Float16* __restrict__ Bt,
                                 const float* __restrict__ bias,
                                 float* __restrict__ C,
                                 int M, int N, int K, int Kp, int act)
{
    const int lane   = threadIdx.x & 31;
    const int wave   = threadIdx.x >> 5;
    const int tilesN = N >> 6;                  // 64-wide column tiles
    const int tilesM = (M + 15) >> 4;
    const int tile   = blockIdx.x * (blockDim.x >> 5) + wave;
    if (tile >= tilesM * tilesN) return;

    const int m0 = (tile / tilesN) << 4;
    const int n0 = (tile % tilesN) << 6;

    const int hl = lane >> 4;                   // wave half (0/1)
    const int lr = lane & 15;
    const int r  = m0 + lr;                     // A row for this lane
    const size_t rowBase = (size_t)((r < M) ? r : (M - 1)) * (size_t)K;

    v8f_t acc[4] = {{}, {}, {}, {}};

    for (int k0 = 0; k0 < Kp; k0 += 32) {
        // ---- A fragment: 16x32 f16; lane = row; element e -> K = (e&7)+(e>=8?16:0)+8*hl
        v16h_t af;
        const int ka = k0 + hl * 8;
        if (k0 + 32 <= K) {                     // full slab: unconditional b128 loads
            const v4f_t* ap = (const v4f_t*)(A + rowBase + ka);
            const v4f_t a0 = ap[0];             // K = ka..ka+3
            const v4f_t a1 = ap[1];             // K = ka+4..ka+7
            const v4f_t a2 = ap[4];             // K = ka+16..ka+19
            const v4f_t a3 = ap[5];             // K = ka+20..ka+23
#pragma unroll
            for (int e = 0; e < 4; ++e) {
                af[e]      = (_Float16)a0[e];
                af[e + 4]  = (_Float16)a1[e];
                af[e + 8]  = (_Float16)a2[e];
                af[e + 12] = (_Float16)a3[e];
            }
        } else {                                // ragged tail (pre-projection K=300 only)
#pragma unroll
            for (int e = 0; e < 8; ++e) {
                const int kk  = ka + e;
                const int kk2 = ka + 16 + e;
                const float v  = A[rowBase + ((kk  < K) ? kk  : (K - 1))];
                const float v2 = A[rowBase + ((kk2 < K) ? kk2 : (K - 1))];
                af[e]     = (kk  < K) ? (_Float16)v  : (_Float16)0.f;
                af[e + 8] = (kk2 < K) ? (_Float16)v2 : (_Float16)0.f;
            }
        }
        if (k0 + 64 <= K)
            __builtin_prefetch(A + rowBase + k0 + 32 + hl * 8, 0, 1);  // global_prefetch_b8

        // ---- 4 column sub-tiles: B fragment from Bt[N,Kp]; element e -> K = (e&7)+(e>=8?16:0)+8*hl
#pragma unroll
        for (int nt = 0; nt < 4; ++nt) {
            const int c = n0 + nt * 16 + lr;
            const v8h_t* bp = (const v8h_t*)(Bt + (size_t)c * Kp + k0 + hl * 8);
            const v8h_t b0 = bp[0];
            const v8h_t b1 = bp[2];
            const v16h_t bf = __builtin_shufflevector(
                b0, b1, 0, 1, 2, 3, 4, 5, 6, 7, 8, 9, 10, 11, 12, 13, 14, 15);
            acc[nt] = __builtin_amdgcn_wmma_f32_16x16x32_f16(
                          false, af, false, bf, (short)0, acc[nt], false, false);
        }
    }

    // ---- epilogue: C/D layout row = m0 + j + 8*hl, col = c
    if (m0 + 16 <= M) {                         // full row tile: unconditional stores
#pragma unroll
        for (int nt = 0; nt < 4; ++nt) {
            const int c = n0 + nt * 16 + lr;
            const float bv = bias ? bias[c] : 0.f;
#pragma unroll
            for (int j = 0; j < 8; ++j) {
                const int rr = m0 + j + hl * 8;
                float v = acc[nt][j] + bv;
                if (act == 1)      v = (v > 0.f) ? v : 0.01f * v;
                else if (act == 2) v = gelu_exact(v);
                C[(size_t)rr * N + c] = v;
            }
        }
    } else {
#pragma unroll
        for (int nt = 0; nt < 4; ++nt) {
            const int c = n0 + nt * 16 + lr;
            const float bv = bias ? bias[c] : 0.f;
#pragma unroll
            for (int j = 0; j < 8; ++j) {
                const int rr = m0 + j + hl * 8;
                float v = acc[nt][j] + bv;
                if (act == 1)      v = (v > 0.f) ? v : 0.01f * v;
                else if (act == 2) v = gelu_exact(v);
                if (rr < M) C[(size_t)rr * N + c] = v;
            }
        }
    }
}

// ---------------------------------------------------------------------------
// Elementwise / GAT helper kernels
// ---------------------------------------------------------------------------

__global__ void bcast_rowvec_kernel(float* __restrict__ buf,
                                    const float* __restrict__ vec,
                                    int N, int D)
{
    const size_t i = (size_t)blockIdx.x * blockDim.x + threadIdx.x;
    if (i < (size_t)N * D) buf[i] = vec[i % D];
}

__global__ void zero_kernel(float* __restrict__ buf, size_t n)
{
    const size_t i = (size_t)blockIdx.x * blockDim.x + threadIdx.x;
    if (i < n) buf[i] = 0.f;
}

// order-preserving float<->uint mapping for atomic max
__device__ __forceinline__ unsigned fkey(float f) {
    unsigned u = __float_as_uint(f);
    return (u & 0x80000000u) ? ~u : (u | 0x80000000u);
}
__device__ __forceinline__ float funkey(unsigned k) {
    unsigned u = (k & 0x80000000u) ? (k & 0x7fffffffu) : ~k;
    return __uint_as_float(u);
}

__global__ void init_node_kernel(unsigned* __restrict__ mkey,
                                 float* __restrict__ denom, int N)
{
    const int n = blockIdx.x * blockDim.x + threadIdx.x;
    if (n < N) {
        mkey[n]  = fkey(-3.0e38f);
        denom[n] = 0.f;
    }
}

// alpha_s[n] = ht[n,:]·as_w ; alpha_d[n] = ht[n,:]·ad_w
__global__ void alpha_kernel(const float* __restrict__ ht,
                             const float* __restrict__ as_w,
                             const float* __restrict__ ad_w,
                             float* __restrict__ alpha_s,
                             float* __restrict__ alpha_d,
                             int N, int D)
{
    const int n = blockIdx.x * blockDim.x + threadIdx.x;
    if (n >= N) return;
    float s = 0.f, d2 = 0.f;
    const float* row = ht + (size_t)n * D;
    for (int d = 0; d < D; ++d) { float v = row[d]; s += v * as_w[d]; d2 += v * ad_w[d]; }
    alpha_s[n] = s; alpha_d[n] = d2;
}

// pass 1: segment max of leaky_relu(alpha_s[src]+alpha_d[dst], 0.2) per dst
__global__ void edge_max_kernel(const int* __restrict__ src,
                                const int* __restrict__ dst,
                                const float* __restrict__ alpha_s,
                                const float* __restrict__ alpha_d,
                                unsigned* __restrict__ mkey,
                                int E, int N)
{
    const int t = blockIdx.x * blockDim.x + threadIdx.x;
    if (t >= E + N) return;
    int s, d;
    if (t < E) { s = src[t]; d = dst[t]; } else { s = d = t - E; }
    float e = alpha_s[s] + alpha_d[d];
    e = (e > 0.f) ? e : 0.2f * e;
    atomicMax(&mkey[d], fkey(e));
}

// pass 2: w = exp(e - m[dst]); segment-sum into denom; stash w
__global__ void edge_exp_kernel(const int* __restrict__ src,
                                const int* __restrict__ dst,
                                const float* __restrict__ alpha_s,
                                const float* __restrict__ alpha_d,
                                const unsigned* __restrict__ mkey,
                                float* __restrict__ denom,
                                float* __restrict__ ew,
                                int E, int N)
{
    const int t = blockIdx.x * blockDim.x + threadIdx.x;
    if (t >= E + N) return;
    int s, d;
    if (t < E) { s = src[t]; d = dst[t]; } else { s = d = t - E; }
    float e = alpha_s[s] + alpha_d[d];
    e = (e > 0.f) ? e : 0.2f * e;
    float w = expf(e - funkey(mkey[d]));
    ew[t] = w;
    atomicAdd(&denom[d], w);
}

// pass 3: xt[dst,:] += ht[src,:] * (w / denom[dst]); one wave32 per edge,
// lanes cover contiguous dims -> coalesced global_atomic_add_f32 bursts.
__global__ void edge_agg_kernel(const int* __restrict__ src,
                                const int* __restrict__ dst,
                                const float* __restrict__ ew,
                                const float* __restrict__ denom,
                                const float* __restrict__ ht,
                                float* __restrict__ xt,
                                int E, int N, int D)
{
    const int lane = threadIdx.x & 31;
    const int e    = blockIdx.x * (blockDim.x >> 5) + (threadIdx.x >> 5);
    if (e >= E + N) return;
    int s, d;
    if (e < E) { s = src[e]; d = dst[e]; } else { s = d = e - E; }
    const float coeff = ew[e] / (denom[d] + 1e-16f);
    const float* hrow = ht + (size_t)s * D;
    float* xrow       = xt + (size_t)d * D;
    for (int d0 = lane; d0 < D; d0 += 32)
        atomicAdd(&xrow[d0], hrow[d0] * coeff);
}

// dots[n*T+t] = Q[n,:]·Kt[n,:]
__global__ void dot_kernel(const float* __restrict__ Q,
                           const float* __restrict__ Kt,
                           float* __restrict__ dots,
                           int N, int D, int T, int t)
{
    const int n = blockIdx.x * blockDim.x + threadIdx.x;
    if (n >= N) return;
    float s = 0.f;
    const float* q = Q  + (size_t)n * D;
    const float* k = Kt + (size_t)n * D;
    for (int d = 0; d < D; ++d) s += q[d] * k[d];
    dots[(size_t)n * T + t] = s;
}

// softmax over T logits per node, scaled by temp[layer]
__global__ void softmax_t_kernel(float* __restrict__ dots,
                                 const float* __restrict__ temp,
                                 int layer, int N, int T)
{
    const int n = blockIdx.x * blockDim.x + threadIdx.x;
    if (n >= N) return;
    const float tv = temp[layer];
    float l[8];
    float m = -3.0e38f;
    for (int t = 0; t < T; ++t) { l[t] = dots[(size_t)n * T + t] * tv; m = fmaxf(m, l[t]); }
    float sum = 0.f;
    for (int t = 0; t < T; ++t) { l[t] = expf(l[t] - m); sum += l[t]; }
    const float inv = 1.f / sum;
    for (int t = 0; t < T; ++t) dots[(size_t)n * T + t] = l[t] * inv;
}

// hatt[n,d] += dots[n*T+t] * Vt[n,d]
__global__ void accumv_kernel(float* __restrict__ hatt,
                              const float* __restrict__ dots,
                              const float* __restrict__ Vt,
                              int N, int D, int T, int t)
{
    const size_t i = (size_t)blockIdx.x * blockDim.x + threadIdx.x;
    if (i >= (size_t)N * D) return;
    const size_t n = i / D;
    hatt[i] += dots[n * T + t] * Vt[i];
}

__device__ __forceinline__ float block_sum(float v, float* sbuf, int d, int D)
{
    sbuf[d] = v;
    __syncthreads();
    for (int s = D >> 1; s > 0; s >>= 1) {
        if (d < s) sbuf[d] += sbuf[d + s];
        __syncthreads();
    }
    float r = sbuf[0];
    __syncthreads();
    return r;
}

// h = LN(h + gelu(hatt), g, b); one block (D threads) per node
__global__ void combine_ln_kernel(float* __restrict__ h,
                                  const float* __restrict__ hatt,
                                  const float* __restrict__ g,
                                  const float* __restrict__ b,
                                  int N, int D)
{
    __shared__ float sbuf[1024];
    const int n = blockIdx.x;
    const int d = threadIdx.x;
    float v = h[(size_t)n * D + d] + gelu_exact(hatt[(size_t)n * D + d]);
    const float mu  = block_sum(v, sbuf, d, D) / (float)D;
    const float dv  = v - mu;
    const float var = block_sum(dv * dv, sbuf, d, D) / (float)D;
    h[(size_t)n * D + d] = dv * rsqrtf(var + 1e-12f) * g[d] + b[d];
}

// h = LN(h + add, g, b); one block (D threads) per node
__global__ void res_ln_kernel(float* __restrict__ h,
                              const float* __restrict__ add,
                              const float* __restrict__ g,
                              const float* __restrict__ b,
                              int N, int D)
{
    __shared__ float sbuf[1024];
    const int n = blockIdx.x;
    const int d = threadIdx.x;
    float v = h[(size_t)n * D + d] + add[(size_t)n * D + d];
    const float mu  = block_sum(v, sbuf, d, D) / (float)D;
    const float dv  = v - mu;
    const float var = block_sum(dv * dv, sbuf, d, D) / (float)D;
    h[(size_t)n * D + d] = dv * rsqrtf(var + 1e-12f) * g[d] + b[d];
}

// ---------------------------------------------------------------------------
// Host-side orchestration
// ---------------------------------------------------------------------------
extern "C" void kernel_launch(void* const* d_in, const int* in_sizes, int n_in,
                              void* d_out, int out_size, void* d_ws, size_t ws_size,
                              hipStream_t stream)
{
    const float* x     = (const float*)d_in[0];
    const int*   ei    = (const int*)  d_in[1];
    const float* W_pre = (const float*)d_in[2];
    const float* b_pre = (const float*)d_in[3];
    const float* W_gat = (const float*)d_in[4];
    const float* att_s = (const float*)d_in[5];
    const float* att_d = (const float*)d_in[6];
    const float* b_gat = (const float*)d_in[7];
    const float* Wq = (const float*)d_in[8];  const float* bq = (const float*)d_in[9];
    const float* Wk = (const float*)d_in[10]; const float* bk = (const float*)d_in[11];
    const float* Wv = (const float*)d_in[12]; const float* bv = (const float*)d_in[13];
    const float* temp = (const float*)d_in[14];
    const float* ln_g = (const float*)d_in[15]; const float* ln_b = (const float*)d_in[16];
    const float* Wm = (const float*)d_in[17]; const float* bm = (const float*)d_in[18];
    const float* Wo = (const float*)d_in[19]; const float* bo = (const float*)d_in[20];
    const float* lnr_g = (const float*)d_in[21]; const float* lnr_b = (const float*)d_in[22];
    const float* W_out = (const float*)d_in[23]; const float* b_out = (const float*)d_in[24];

    // Derive shapes from input sizes
    const int D    = in_sizes[3];                       // 128
    const int L    = in_sizes[14];                      // 2
    const int Din  = in_sizes[2] / D;                   // 300
    const int N    = in_sizes[0] / Din;                 // 50000
    const int T    = in_sizes[4] / (L * D * D);         // 3
    const int E    = in_sizes[1] / (2 * T);             // 300000
    const int M    = in_sizes[18] / L;                  // 512
    const int Dout = in_sizes[24];                      // 128

    float* ws = (float*)d_ws;
    const size_t ND = (size_t)N * D;

    float* h    = ws;                 // [N,D]  node features (persistent)
    float* Q    = ws + 1 * ND;        // [N,D]
    float* hatt = ws + 2 * ND;        // [N,D]
    float* tmp  = ws + 3 * ND;        // [N,D]  per-type ht / K_t / V_t scratch
    float* xt   = ws + 4 * ND;        // [T,N,D] GAT outputs
    float* mid  = ws + 2 * ND;        // [N,M]  aliases hatt/tmp/xt[0..1] (dead at MLP time)
    float* mlpo = ws + 2 * ND + (size_t)N * M;   // [N,D] right after mid (inside dead xt)
    float* dots = ws + (size_t)(4 + T) * ND;     // [N,T]
    float* a_s  = dots + (size_t)T * N;          // [N]
    float* a_d  = a_s + N;                       // [N]
    unsigned* mk = (unsigned*)(a_d + N);         // [N]
    float* denom = (float*)(mk + N);             // [N]
    float* ew    = denom + N;                    // [E+N]
    // half B^T staging buffer (16-byte aligned)
    size_t btOff = (size_t)((ew + (E + N)) - ws);
    btOff = (btOff + 7) & ~(size_t)7;
    _Float16* Bt = (_Float16*)(ws + btOff);

    const int Etot = E + N;
    const int TB = 256;
    auto blocks1 = [&](size_t n) { return (unsigned)((n + TB - 1) / TB); };

    auto gemm = [&](const float* A, const float* B, const float* bias, float* C,
                    int m, int n, int k, int act) {
        const int kp = (k + 31) & ~31;
        convert_bt_kernel<<<blocks1((size_t)n * kp), TB, 0, stream>>>(B, Bt, k, n, kp);
        const int tiles = ((m + 15) / 16) * (n / 64);
        const int nBlk  = (tiles + (TB / 32) - 1) / (TB / 32);
        wmma_gemm_kernel<<<nBlk, TB, 0, stream>>>(A, Bt, bias, C, m, n, k, kp, act);
    };

    // ---- pre-projection: h = leaky_relu(x @ W_pre + b_pre, 0.01)
    gemm(x, W_pre, b_pre, h, N, D, Din, /*act=*/1);

    for (int i = 0; i < L; ++i) {
        // ---------------- per-type GATConv ----------------
        for (int t = 0; t < T; ++t) {
            const int* srcp = ei + (size_t)t * 2 * E;
            const int* dstp = srcp + E;
            const float* Wg = W_gat + ((size_t)i * T + t) * D * D;
            const float* as = att_s + ((size_t)i * T + t) * D;
            const float* ad = att_d + ((size_t)i * T + t) * D;
            const float* bg = b_gat + ((size_t)i * T + t) * D;
            float* xt_t = xt + (size_t)t * ND;

            gemm(h, Wg, nullptr, tmp, N, D, D, 0);                           // ht = h @ W
            alpha_kernel<<<blocks1(N), TB, 0, stream>>>(tmp, as, ad, a_s, a_d, N, D);
            bcast_rowvec_kernel<<<blocks1(ND), TB, 0, stream>>>(xt_t, bg, N, D);
            init_node_kernel<<<blocks1(N), TB, 0, stream>>>(mk, denom, N);
            edge_max_kernel<<<blocks1(Etot), TB, 0, stream>>>(srcp, dstp, a_s, a_d, mk, E, N);
            edge_exp_kernel<<<blocks1(Etot), TB, 0, stream>>>(srcp, dstp, a_s, a_d, mk, denom, ew, E, N);
            edge_agg_kernel<<<(Etot + (TB / 32) - 1) / (TB / 32), TB, 0, stream>>>(
                srcp, dstp, ew, denom, tmp, xt_t, E, N, D);
        }

        // ---------------- type attention ----------------
        gemm(h, Wq + (size_t)i * D * D, bq + (size_t)i * D, Q, N, D, D, 0);  // Q
        for (int t = 0; t < T; ++t) {
            gemm(xt + (size_t)t * ND, Wk + (size_t)i * D * D, bk + (size_t)i * D,
                 tmp, N, D, D, 0);                                           // K_t
            dot_kernel<<<blocks1(N), TB, 0, stream>>>(Q, tmp, dots, N, D, T, t);
        }
        softmax_t_kernel<<<blocks1(N), TB, 0, stream>>>(dots, temp, i, N, T);
        zero_kernel<<<blocks1(ND), TB, 0, stream>>>(hatt, ND);
        for (int t = 0; t < T; ++t) {
            gemm(xt + (size_t)t * ND, Wv + (size_t)i * D * D, bv + (size_t)i * D,
                 tmp, N, D, D, 0);                                           // V_t
            accumv_kernel<<<blocks1(ND), TB, 0, stream>>>(hatt, dots, tmp, N, D, T, t);
        }
        combine_ln_kernel<<<N, D, 0, stream>>>(h, hatt, ln_g, ln_b, N, D);

        // ---------------- MLP block ----------------
        gemm(h, Wm + (size_t)i * D * M, bm + (size_t)i * M, mid, N, M, D, /*act=*/2);
        gemm(mid, Wo + (size_t)i * M * D, bo + (size_t)i * D, mlpo, N, D, M, 0);
        res_ln_kernel<<<N, D, 0, stream>>>(h, mlpo,
                                           lnr_g + (size_t)i * D, lnr_b + (size_t)i * D, N, D);
    }

    // ---- output head: out = gelu(h @ W_out + b_out)
    gemm(h, W_out, b_out, (float*)d_out, N, Dout, D, /*act=*/2);
}
